// FSNet_63161789055042
// MI455X (gfx1250) — matrix operand order repaired and checked by reading, not compiled, not run on hardware
//
#include <hip/hip_runtime.h>
#include <hip/hip_bf16.h>

// ---------------------------------------------------------------------------
// FSNet on MI455X (gfx1250): bf16 WMMA GEMMs (register-blocked) +
// LDS-resident GRU scans (bank-conflict-padded, no scratch spills).
// ---------------------------------------------------------------------------

typedef __attribute__((ext_vector_type(16))) __bf16 v16bf;
typedef __attribute__((ext_vector_type(8)))  float  v8f;

union FragBF {
    v16bf v;
    uint4 q[2];
};

__device__ __forceinline__ unsigned short f2bf(float f) {
    unsigned int u = __float_as_uint(f);
    unsigned int r = u + 0x7FFFu + ((u >> 16) & 1u);   // round-to-nearest-even
    return (unsigned short)(r >> 16);
}

__device__ __forceinline__ v8f wmma_bf16(const FragBF& a, const FragBF& b, v8f c) {
    // D = A(16x32 bf16) * B(32x16 bf16) + C(16x16 f32)
    return __builtin_amdgcn_wmma_f32_16x16x32_bf16(
        /*neg_a=*/false, a.v, /*neg_b=*/false, b.v,
        /*c_mod=*/(short)0, c, /*reuse_a=*/false, /*reuse_b=*/false);
}

// Fast activations: v_exp_f32 + v_rcp_f32 (avoid IEEE-divide expansion).
__device__ __forceinline__ float fast_sigmoid(float x) {
    return __builtin_amdgcn_rcpf(1.0f + __expf(-x));
}
__device__ __forceinline__ float fast_tanh(float x) {
    return 2.0f * __builtin_amdgcn_rcpf(1.0f + __expf(-2.0f * x)) - 1.0f;
}

// ---------------------------------------------------------------------------
// Elementwise f32 -> bf16 convert
// ---------------------------------------------------------------------------
__global__ void k_f2bf(const float* __restrict__ in, unsigned short* __restrict__ out, int n) {
    int i = blockIdx.x * blockDim.x + threadIdx.x;
    if (i < n) out[i] = f2bf(in[i]);
}

// ---------------------------------------------------------------------------
// Embedding gather -> time-major bf16 (row = t*64 + b, 128 cols)
// ---------------------------------------------------------------------------
__global__ void k_embed(const int* __restrict__ x, const float* __restrict__ emb,
                        unsigned short* __restrict__ e) {
    int i = blockIdx.x * blockDim.x + threadIdx.x;   // over 6400*128
    if (i >= 6400 * 128) return;
    int r = i >> 7;            // time-major row: t*64 + b
    int c = i & 127;
    int t = r >> 6;
    int b = r & 63;
    int xi = x[b * 100 + t];
    if (xi > 9999) xi = 9999;
    if (xi < 0) xi = 0;
    e[i] = f2bf(emb[(size_t)xi * 128 + c]);
}

// ---------------------------------------------------------------------------
// Register-blocked bf16 WMMA GEMM: C(M,N) f32 = A(M,K) * W(N,K)^T + bias(N)
// A row-major bf16; W row-major (N,K) bf16 (B-columns contiguous).
// Each wave computes an MB x NB block of 16x16 tiles (A frags reused across
// NB, B frags across MB). Requires M % (16*MB) == 0, N % (16*NB) == 0,
// K % 32 == 0. NT=true streams C with nontemporal stores.
// ---------------------------------------------------------------------------
template <int MB, int NB, bool NT>
__launch_bounds__(256)
__global__ void k_gemm_bf16(const unsigned short* __restrict__ A,
                            const unsigned short* __restrict__ W,
                            const float* __restrict__ bias,
                            float* __restrict__ C,
                            int M, int N, int K, int ldc) {
    const int lane = threadIdx.x & 31;
    const int wave = threadIdx.x >> 5;
    const int nBlkN = N / (16 * NB);
    const int nBlk  = (M / (16 * MB)) * nBlkN;
    const int wid = blockIdx.x * (blockDim.x >> 5) + wave;
    const int nw  = gridDim.x * (blockDim.x >> 5);
    const int colg = lane & 15;
    const int hi8  = (lane < 16) ? 0 : 8;
    const int aoff = (lane < 16) ? 0 : 8;    // A K-group offset (documented layout)
    const int boff = (lane < 16) ? 0 : 8;    // B K-group offset

    for (int blk = wid; blk < nBlk; blk += nw) {
        const int mBase = (blk / nBlkN) * (16 * MB);
        const int nBase = (blk % nBlkN) * (16 * NB);

        v8f acc[MB][NB];
#pragma unroll
        for (int i = 0; i < MB; ++i)
#pragma unroll
            for (int j = 0; j < NB; ++j) acc[i][j] = {};

        const unsigned short* arow[MB];
        const unsigned short* wrow[NB];
#pragma unroll
        for (int i = 0; i < MB; ++i)
            arow[i] = A + (size_t)(mBase + i * 16 + colg) * K + aoff;
#pragma unroll
        for (int j = 0; j < NB; ++j)
            wrow[j] = W + (size_t)(nBase + j * 16 + colg) * K + boff;

        for (int kb = 0; kb < K; kb += 32) {
            FragBF a[MB], b[NB];
#pragma unroll
            for (int i = 0; i < MB; ++i) {
                a[i].q[0] = *(const uint4*)(arow[i] + kb);
                a[i].q[1] = *(const uint4*)(arow[i] + kb + 16);
            }
#pragma unroll
            for (int j = 0; j < NB; ++j) {
                b[j].q[0] = *(const uint4*)(wrow[j] + kb);
                b[j].q[1] = *(const uint4*)(wrow[j] + kb + 16);
            }
#pragma unroll
            for (int i = 0; i < MB; ++i)
#pragma unroll
                for (int j = 0; j < NB; ++j)
                    acc[i][j] = wmma_bf16(a[i], b[j], acc[i][j]);
        }

#pragma unroll
        for (int i = 0; i < MB; ++i) {
#pragma unroll
            for (int j = 0; j < NB; ++j) {
                const int n0 = nBase + j * 16;
                const float bv = bias ? bias[n0 + colg] : 0.0f;
                float* crow = C + (size_t)(mBase + i * 16 + hi8) * ldc + n0 + colg;
#pragma unroll
                for (int r = 0; r < 8; ++r) {
                    const float v = acc[i][j][r] + bv;
                    if (NT) __builtin_nontemporal_store(v, crow + (size_t)r * ldc);
                    else    crow[(size_t)r * ldc] = v;
                }
            }
        }
    }
}

// ---------------------------------------------------------------------------
// GRU scan: gridDim.x = 2 (dir 0 fwd, dir 1 bwd). blockDim = 256 (8 waves).
// LDS layout (dynamic, bank-conflict padded):
//   hF  : f32  hidden state, pitch 132 (rows stride 528B -> bank +4/row)
//   hB  : bf16 hidden copy,  pitch 136 (rows stride 272B -> bank +4/row)
//   wL  : bf16 Whh staged,   pitch 136, 384 rows
// gi: (rows,768) f32, row = tt*gts + b, per-dir col offset dir*384 (has bih).
// y: bf16, 256 cols, row = tt*yt + b*yb, col = dir*128 + j.
// hT: f32 (64,512) base; writes cols [hcol + dir*128, +128).
// ---------------------------------------------------------------------------
#define HF_PITCH 132
#define HB_PITCH 136
#define WL_PITCH 136
#define SMEM_HF_BYTES (64 * HF_PITCH * 4)             // 33792
#define SMEM_HB_BYTES (64 * HB_PITCH * 2)             // 17408
#define SMEM_WL_BYTES (384 * WL_PITCH * 2)            // 104448
#define SMEM_TOTAL_BYTES (SMEM_HF_BYTES + SMEM_HB_BYTES + SMEM_WL_BYTES)

__launch_bounds__(256)
__global__ void k_gru_scan(const float* __restrict__ gi, int gts,
                           const unsigned short* __restrict__ whh_bf16,
                           const float* __restrict__ bhh,
                           unsigned short* __restrict__ y, int yt, int yb,
                           float* __restrict__ hT, int hcol) {
    const int dir  = blockIdx.x;
    const int tid  = threadIdx.x;
    const int lane = tid & 31;
    const int wave = tid >> 5;

    extern __shared__ char smem[];
    float*          hF = (float*)smem;
    unsigned short* hB = (unsigned short*)(smem + SMEM_HF_BYTES);
    unsigned short* wL = (unsigned short*)(smem + SMEM_HF_BYTES + SMEM_HB_BYTES);

    // init h = 0 and stage Whh (this dir) into padded LDS
    for (int i = tid; i < 64 * HF_PITCH; i += 256) hF[i] = 0.0f;
    for (int i = tid; i < 64 * HB_PITCH; i += 256) hB[i] = 0;
    {
        const unsigned short* src = whh_bf16 + (size_t)dir * 384 * 128;
        for (int i = tid; i < 384 * 128; i += 256) {
            int r = i >> 7, c = i & 127;
            wL[r * WL_PITCH + c] = src[i];
        }
    }
    __syncthreads();

    const float* bh = bhh + dir * 384;
    const int colg = lane & 15;
    const int hi8  = (lane < 16) ? 0 : 8;
    const int aoff = (lane < 16) ? 0 : 8;
    const int boff = (lane < 16) ? 0 : 8;

    for (int t = 0; t < 100; ++t) {
        // Compiler barrier: prevent hoisting the (loop-invariant) fragment
        // loads out of the scan loop -> avoids 384-VGPR scratch spills.
        asm volatile("" ::: "memory");
        const int tt = dir ? (99 - t) : t;
        float hnew[4][8];

#pragma unroll
        for (int p = 0; p < 4; ++p) {
            const int q  = wave * 4 + p;      // 32 (m,j) pairs over 8 waves
            const int m0 = (q >> 3) * 16;     // batch-tile
            const int j0 = (q & 7) * 16;      // hidden-tile
            v8f accr = {}, accz = {}, accn = {};
#pragma unroll
            for (int kb = 0; kb < 128; kb += 32) {
                FragBF a, b0, b1, b2;
                const unsigned short* pa = hB + (size_t)(m0 + colg) * HB_PITCH + kb + aoff;
                a.q[0] = *(const uint4*)(pa);
                a.q[1] = *(const uint4*)(pa + 16);
                const unsigned short* pr = wL + (size_t)(0 * 128 + j0 + colg) * WL_PITCH + kb + boff;
                const unsigned short* pz = wL + (size_t)(1 * 128 + j0 + colg) * WL_PITCH + kb + boff;
                const unsigned short* pn = wL + (size_t)(2 * 128 + j0 + colg) * WL_PITCH + kb + boff;
                b0.q[0] = *(const uint4*)(pr); b0.q[1] = *(const uint4*)(pr + 16);
                b1.q[0] = *(const uint4*)(pz); b1.q[1] = *(const uint4*)(pz + 16);
                b2.q[0] = *(const uint4*)(pn); b2.q[1] = *(const uint4*)(pn + 16);
                accr = wmma_bf16(a, b0, accr);
                accz = wmma_bf16(a, b1, accz);
                accn = wmma_bf16(a, b2, accn);
            }
            // Gate math (elementwise in the C/D register layout).
            const int jc  = j0 + colg;
            const float bhr = bh[jc];
            const float bhz = bh[128 + jc];
            const float bhn = bh[256 + jc];
#pragma unroll
            for (int r = 0; r < 8; ++r) {
                const int b = m0 + r + hi8;
                const float* gib = gi + (size_t)(tt * gts + b) * 768 + dir * 384;
                const float ir = gib[jc];
                const float iz = gib[128 + jc];
                const float in = gib[256 + jc];
                const float rg = fast_sigmoid(ir + accr[r] + bhr);
                const float zg = fast_sigmoid(iz + accz[r] + bhz);
                const float ng = fast_tanh(in + rg * (accn[r] + bhn));
                hnew[p][r] = (1.0f - zg) * ng + zg * hF[b * HF_PITCH + jc];
            }
        }
        __syncthreads();   // all reads of h done before overwrites
#pragma unroll
        for (int p = 0; p < 4; ++p) {
            const int q  = wave * 4 + p;
            const int m0 = (q >> 3) * 16;
            const int jc = (q & 7) * 16 + colg;
#pragma unroll
            for (int r = 0; r < 8; ++r) {
                const int b = m0 + r + hi8;
                const float v = hnew[p][r];
                const unsigned short us = f2bf(v);
                hF[b * HF_PITCH + jc] = v;
                hB[b * HB_PITCH + jc] = us;
                y[((size_t)tt * yt + (size_t)b * yb) * 256 + dir * 128 + jc] = us;
            }
        }
        __syncthreads();
    }

    // Final hidden state out.
    for (int i = tid; i < 64 * 128; i += 256) {
        const int b = i >> 7;
        const int c = i & 127;
        hT[(size_t)b * 512 + hcol + dir * 128 + c] = hF[b * HF_PITCH + c];
    }
}

// ---------------------------------------------------------------------------
// fc head: out(64,20) = [encH, decH](64,1024) @ fc_W(20,1024)^T + fc_b
// ---------------------------------------------------------------------------
__global__ void k_fc(const float* __restrict__ encH, const float* __restrict__ decH,
                     const float* __restrict__ W, const float* __restrict__ bias,
                     float* __restrict__ out) {
    int o = blockIdx.x * blockDim.x + threadIdx.x;
    if (o >= 64 * 20) return;
    const int b = o / 20, n = o % 20;
    const float* w = W + (size_t)n * 1024;
    const float* e = encH + (size_t)b * 512;
    const float* d = decH + (size_t)b * 512;
    float s = bias[n];
    for (int k = 0; k < 512; ++k) s += e[k] * w[k];
    for (int k = 0; k < 512; ++k) s += d[k] * w[512 + k];
    out[o] = s;
}

// ---------------------------------------------------------------------------
// Host-side orchestration
// ---------------------------------------------------------------------------
static inline size_t alignUp(size_t v, size_t a) { return (v + a - 1) & ~(a - 1); }

extern "C" void kernel_launch(void* const* d_in, const int* in_sizes, int n_in,
                              void* d_out, int out_size, void* d_ws, size_t ws_size,
                              hipStream_t stream) {
    (void)in_sizes; (void)n_in; (void)out_size; (void)ws_size;

    const int*   x        = (const int*)  d_in[0];
    const float* emb      = (const float*)d_in[1];
    const float* encWih0  = (const float*)d_in[2];
    const float* encWhh0  = (const float*)d_in[3];
    const float* encBih0  = (const float*)d_in[4];
    const float* encBhh0  = (const float*)d_in[5];
    const float* encWih1  = (const float*)d_in[6];
    const float* encWhh1  = (const float*)d_in[7];
    const float* encBih1  = (const float*)d_in[8];
    const float* encBhh1  = (const float*)d_in[9];
    const float* decWih0  = (const float*)d_in[10];
    const float* decWhh0  = (const float*)d_in[11];
    const float* decBih0  = (const float*)d_in[12];
    const float* decBhh0  = (const float*)d_in[13];
    const float* decWih1  = (const float*)d_in[14];
    const float* decWhh1  = (const float*)d_in[15];
    const float* decBih1  = (const float*)d_in[16];
    const float* decBhh1  = (const float*)d_in[17];
    const float* fcW      = (const float*)d_in[18];
    const float* fcB      = (const float*)d_in[19];
    const float* recW     = (const float*)d_in[20];
    const float* recB     = (const float*)d_in[21];

    float* outp = (float*)d_out;           // [0,1280): out ; [1280, +64e6): rec

    // ---- workspace bump allocator ----
    char* ws = (char*)d_ws;
    size_t off = 0;
    auto alloc = [&](size_t bytes) -> void* {
        void* p = ws + off;
        off = alignUp(off + bytes, 256);
        return p;
    };
    unsigned short* eBf      = (unsigned short*)alloc((size_t)6400 * 128 * 2);
    unsigned short* encWih0b = (unsigned short*)alloc((size_t)768 * 128 * 2);
    unsigned short* encWhh0b = (unsigned short*)alloc((size_t)768 * 128 * 2);
    unsigned short* encWih1b = (unsigned short*)alloc((size_t)768 * 256 * 2);
    unsigned short* encWhh1b = (unsigned short*)alloc((size_t)768 * 128 * 2);
    unsigned short* decWih0b = (unsigned short*)alloc((size_t)768 * 512 * 2);
    unsigned short* decWhh0b = (unsigned short*)alloc((size_t)768 * 128 * 2);
    unsigned short* decWih1b = (unsigned short*)alloc((size_t)768 * 256 * 2);
    unsigned short* decWhh1b = (unsigned short*)alloc((size_t)768 * 128 * 2);
    unsigned short* recWb    = (unsigned short*)alloc((size_t)10000 * 256 * 2);
    float*          gi       = (float*)alloc((size_t)6400 * 768 * 4);
    float*          giDec0   = (float*)alloc((size_t)64 * 768 * 4);
    unsigned short* y0       = (unsigned short*)alloc((size_t)6400 * 256 * 2); // time-major
    unsigned short* yrec     = (unsigned short*)alloc((size_t)6400 * 256 * 2); // batch-major
    float*          encH     = (float*)alloc((size_t)64 * 512 * 4);
    float*          decH     = (float*)alloc((size_t)64 * 512 * 4);
    unsigned short* encHb    = (unsigned short*)alloc((size_t)64 * 512 * 2);

    auto cvt = [&](const float* src, unsigned short* dst, int n) {
        k_f2bf<<<(n + 255) / 256, 256, 0, stream>>>(src, dst, n);
    };
    // gi GEMMs: 2x4 tile blocks (N=768 = 48 tiles, divisible by 4)
    auto gemm24 = [&](const unsigned short* A, const unsigned short* W, const float* bias,
                      float* C, int M, int N, int K, int ldc) {
        int nBlk   = (M / 32) * (N / 64);
        int blocks = (nBlk + 7) / 8;
        if (blocks > 4096) blocks = 4096;
        if (blocks < 1) blocks = 1;
        k_gemm_bf16<2, 4, false><<<blocks, 256, 0, stream>>>(A, W, bias, C, M, N, K, ldc);
    };
    // rec GEMM: 2x5 tile blocks (N=10000 = 625 tiles = 125*5), NT streaming C
    auto gemm25nt = [&](const unsigned short* A, const unsigned short* W, const float* bias,
                        float* C, int M, int N, int K, int ldc) {
        int nBlk   = (M / 32) * (N / 80);
        int blocks = (nBlk + 7) / 8;
        if (blocks > 4096) blocks = 4096;
        if (blocks < 1) blocks = 1;
        k_gemm_bf16<2, 5, true><<<blocks, 256, 0, stream>>>(A, W, bias, C, M, N, K, ldc);
    };
    auto scan = [&](const float* giPtr, int gts, const unsigned short* whh, const float* bhhPtr,
                    unsigned short* yPtr, int yt, int yb, float* hTPtr, int hcol) {
        k_gru_scan<<<2, 256, SMEM_TOTAL_BYTES, stream>>>(giPtr, gts, whh, bhhPtr,
                                                         yPtr, yt, yb, hTPtr, hcol);
    };

    // --- weight conversions (f32 -> bf16, keep (N,K) row-major layout) ---
    cvt(encWih0, encWih0b, 768 * 128);
    cvt(encWhh0, encWhh0b, 768 * 128);
    cvt(encWih1, encWih1b, 768 * 256);
    cvt(encWhh1, encWhh1b, 768 * 128);
    cvt(decWih0, decWih0b, 768 * 512);
    cvt(decWhh0, decWhh0b, 768 * 128);
    cvt(decWih1, decWih1b, 768 * 256);
    cvt(decWhh1, decWhh1b, 768 * 128);
    cvt(recW,    recWb,    10000 * 256);

    // --- encoder ---
    k_embed<<<(6400 * 128 + 255) / 256, 256, 0, stream>>>(x, emb, eBf);

    gemm24(eBf, encWih0b, encBih0, gi, 6400, 768, 128, 768);                  // gi enc L0
    scan(gi, 64, encWhh0b, encBhh0, y0, 64, 1, encH, 0);                      // enc L0 scan
    gemm24(y0, encWih1b, encBih1, gi, 6400, 768, 256, 768);                   // gi enc L1
    scan(gi, 64, encWhh1b, encBhh1, y0, 64, 1, encH, 256);                    // enc L1 scan

    // --- decoder (input = enc_h broadcast over T: gts = 0) ---
    cvt(encH, encHb, 64 * 512);
    gemm24(encHb, decWih0b, decBih0, giDec0, 64, 768, 512, 768);              // gi dec L0 (t-invariant)
    scan(giDec0, 0, decWhh0b, decBhh0, y0, 64, 1, decH, 0);                   // dec L0 scan
    gemm24(y0, decWih1b, decBih1, gi, 6400, 768, 256, 768);                   // gi dec L1
    scan(gi, 64, decWhh1b, decBhh1, yrec, 1, 100, decH, 256);                 // dec L1 scan (batch-major y)

    // --- heads ---
    gemm25nt(yrec, recWb, recB, outp + 1280, 6400, 10000, 256, 10000);        // rec (big GEMM)
    k_fc<<<(1280 + 255) / 256, 256, 0, stream>>>(encH, decH, fcW, fcB, outp); // out
}
